// GraphAttentionLayer_8418135900363
// MI455X (gfx1250) — matrix operand order repaired and checked by reading, not compiled, not run on hardware
//
#include <hip/hip_runtime.h>
#include <hip/hip_bf16.h>

// ---------------- problem constants (from reference) ----------------
#define N_NODES 100000
#define N_EDGES 1600000
#define IN_DIM  256
#define OUT_DIM 128
#define M_TILES (N_NODES / 16)      // 6250 (exact)

// ---------------- workspace layout (bytes, all 256-aligned) ---------
#define OFF_H       0ull                      // 100000*128 f32 = 51,200,000 B
#define OFF_WPACK   51200000ull               // 8*8*32*16 bf16 = 65,536 B
#define OFF_SSRC    51265536ull               // 100000 f32 (padded 400,128 B)
#define OFF_SDST    51665664ull
#define OFF_SEGMAX  52065792ull
#define OFF_DENOM   52465920ull
#define OFF_EARR    52866048ull               // 1.6M f32 = 6,400,000 B
// total ~59.3 MB

typedef __attribute__((ext_vector_type(16))) __bf16 v16bf;
typedef __attribute__((ext_vector_type(8)))  float  v8f;

// ---------------- W repack: fragment-major bf16 ---------------------
// B-fragment layout (16-bit B 32x16, wave32): lane L holds col N = L%16,
// elements e=0..7 hold K = (L/16)*8 + e, e=8..15 hold K = 16 + (L/16)*8 + (e-8) within the 32-K tile.
// wpack linear index: (((ntile*8 + ktile)*32 + lane)*16 + e)
__global__ __launch_bounds__(256) void gat_pack_w(const float* __restrict__ W,
                                                  __bf16* __restrict__ wpack) {
    int i = blockIdx.x * 256 + threadIdx.x;      // 32768 total
    int e    = i & 15;
    int lane = (i >> 4) & 31;
    int k    = (i >> 9) & 7;
    int n    = i >> 12;
    int K = k * 32 + (e >> 3) * 16 + (lane >> 4) * 8 + (e & 7);
    int N = n * 16 + (lane & 15);
    wpack[i] = (__bf16)W[K * OUT_DIM + N];
}

// ---------------- init: zero accumulator, segmax=-inf, denom=0 ------
__global__ __launch_bounds__(256) void gat_init(float* __restrict__ out,
                                                float* __restrict__ segmax,
                                                float* __restrict__ denom) {
    size_t i = (size_t)blockIdx.x * 256 + threadIdx.x;
    if (i < (size_t)N_NODES * OUT_DIM) out[i] = 0.0f;
    if (i < N_NODES) {
        segmax[i] = __int_as_float(0xFF800000);  // -inf
        denom[i]  = 0.0f;
    }
}

// ---------------- GEMM: h = x @ W (bf16 WMMA), fused s_src/s_dst ----
// Block = 16 rows x 128 cols. 8 waves; wave w owns cols [16w, 16w+16).
__global__ __launch_bounds__(256) void gat_gemm(const float* __restrict__ x,
                                                const __bf16* __restrict__ wpack,
                                                const float* __restrict__ a,
                                                float* __restrict__ h,
                                                float* __restrict__ ssrc,
                                                float* __restrict__ sdst) {
    __shared__ __bf16 Atile[16 * IN_DIM];   // 8 KB
    __shared__ float  sred[2][16];
    const int tid = threadIdx.x;
    const size_t m0 = (size_t)blockIdx.x * 16;

    if (tid < 32) sred[tid >> 4][tid & 15] = 0.0f;

    // stage A tile: 16 rows x 256 K, fp32 -> bf16 into LDS
    {
        int row = tid >> 4, seg = tid & 15;
        const float4* xp = (const float4*)(x + (m0 + row) * IN_DIM + seg * 16);
        float4 vv[4];
#pragma unroll
        for (int q = 0; q < 4; ++q) vv[q] = xp[q];
        __bf16* dp = Atile + row * IN_DIM + seg * 16;
#pragma unroll
        for (int q = 0; q < 4; ++q) {
            dp[q * 4 + 0] = (__bf16)vv[q].x;
            dp[q * 4 + 1] = (__bf16)vv[q].y;
            dp[q * 4 + 2] = (__bf16)vv[q].z;
            dp[q * 4 + 3] = (__bf16)vv[q].w;
        }
    }
    __syncthreads();

    const int wave = tid >> 5;
    const int lane = tid & 31;
    const int row  = lane & 15;   // A: M index / B,C,D: N index within tile
    const int half = lane >> 4;

    v8f c = {};
#pragma unroll
    for (int k = 0; k < 8; ++k) {
        // A fragment (16x32 bf16): elems 0..7 -> K = half*8 + j,
        //                          elems 8..15 -> K = 16 + half*8 + j
        v16bf afrag;
        const __bf16* ap = Atile + row * IN_DIM + k * 32 + half * 8;
#pragma unroll
        for (int j = 0; j < 8; ++j) {
            afrag[j]     = ap[j];        // 16B contiguous -> ds_load_b128
            afrag[8 + j] = ap[16 + j];
        }
        // B fragment: 32 contiguous bytes per lane from the packed layout
        v16bf bfrag = *(const v16bf*)(wpack + (((size_t)(wave * 8 + k) * 32 + lane) << 4));
        c = __builtin_amdgcn_wmma_f32_16x16x32_bf16(
                false, afrag, false, bfrag, (short)0, c, false, false);
    }

    // store h; D layout: VGPR r -> M = r + 8*half, N = row (+ wave tile)
    const int col = wave * 16 + row;
#pragma unroll
    for (int r = 0; r < 8; ++r) {
        h[(m0 + half * 8 + r) * OUT_DIM + col] = c[r];
    }

    // fused epilogue: s_src[m] = sum_n h[m,n]*a[n]; s_dst with a[128+n]
    const float asrc = a[col];
    const float adst = a[OUT_DIM + col];
#pragma unroll
    for (int r = 0; r < 8; ++r) {
        float vs = c[r] * asrc;
        float vd = c[r] * adst;
#pragma unroll
        for (int m = 1; m < 16; m <<= 1) {     // butterfly within 16-lane half
            vs += __shfl_xor(vs, m, 32);
            vd += __shfl_xor(vd, m, 32);
        }
        if (row == 0) {
            int mrow = half * 8 + r;
            atomicAdd(&sred[0][mrow], vs);     // ds_add_f32
            atomicAdd(&sred[1][mrow], vd);
        }
    }
    __syncthreads();
    if (tid < 16)       ssrc[m0 + tid]        = sred[0][tid];
    else if (tid < 32)  sdst[m0 + (tid - 16)] = sred[1][tid - 16];
}

// ---------------- float atomic max via int/uint punning -------------
__device__ __forceinline__ void atomicMaxF(float* addr, float v) {
    if (v >= 0.0f) atomicMax((int*)addr, __float_as_int(v));
    else           atomicMin((unsigned int*)addr, (unsigned int)__float_as_int(v));
}

// ---------------- edge scores + segment max -------------------------
__global__ __launch_bounds__(256) void gat_edge_score(const long long* __restrict__ src,
                                                      const long long* __restrict__ dst,
                                                      const float* __restrict__ ssrc,
                                                      const float* __restrict__ sdst,
                                                      float* __restrict__ eArr,
                                                      float* __restrict__ segmax) {
    int i = blockIdx.x * 256 + threadIdx.x;
    if (i >= N_EDGES) return;
    long long s = src[i], d = dst[i];
    float e = ssrc[s] + sdst[d];
    e = e > 0.0f ? e : 0.2f * e;          // leaky_relu(0.2)
    eArr[i] = e;
    atomicMaxF(segmax + s, e);
}

// ---------------- weighted scatter-aggregate (1 wave per edge) ------
__global__ __launch_bounds__(256) void gat_aggregate(const long long* __restrict__ src,
                                                     const long long* __restrict__ dst,
                                                     const float* __restrict__ eArr,
                                                     const float* __restrict__ segmax,
                                                     const float* __restrict__ h,
                                                     float* __restrict__ denom,
                                                     float* __restrict__ out) {
    int edge = blockIdx.x * 8 + (threadIdx.x >> 5);
    int lane = threadIdx.x & 31;
    if (edge >= N_EDGES) return;
    long long s = src[edge], d = dst[edge];
    float w = __expf(eArr[edge] - segmax[s]);
    if (lane == 0) atomicAdd(denom + s, w);
    float4 hv = *(const float4*)(h + (size_t)d * OUT_DIM + lane * 4);
    float* op = out + (size_t)s * OUT_DIM + lane * 4;
    atomicAdd(op + 0, w * hv.x);
    atomicAdd(op + 1, w * hv.y);
    atomicAdd(op + 2, w * hv.z);
    atomicAdd(op + 3, w * hv.w);
}

// ---------------- normalize + ELU in place --------------------------
__global__ __launch_bounds__(256) void gat_finalize(float* __restrict__ out,
                                                    const float* __restrict__ denom) {
    size_t i = (size_t)blockIdx.x * 256 + threadIdx.x;
    if (i >= (size_t)N_NODES * OUT_DIM) return;
    float dn = denom[i >> 7];
    float v = out[i] / (dn > 0.0f ? dn : 1.0f);
    out[i] = v > 0.0f ? v : (__expf(v) - 1.0f);
}

extern "C" void kernel_launch(void* const* d_in, const int* in_sizes, int n_in,
                              void* d_out, int out_size, void* d_ws, size_t ws_size,
                              hipStream_t stream) {
    (void)in_sizes; (void)n_in; (void)out_size; (void)ws_size;
    const float*     x     = (const float*)d_in[0];
    const long long* edges = (const long long*)d_in[1];   // int64 per reference
    const float*     W     = (const float*)d_in[2];
    const float*     a     = (const float*)d_in[3];
    const long long* src   = edges;
    const long long* dst   = edges + N_EDGES;

    char* ws = (char*)d_ws;
    float*  h      = (float*)(ws + OFF_H);
    __bf16* wpack  = (__bf16*)(ws + OFF_WPACK);
    float*  ssrc   = (float*)(ws + OFF_SSRC);
    float*  sdst   = (float*)(ws + OFF_SDST);
    float*  segmax = (float*)(ws + OFF_SEGMAX);
    float*  denom  = (float*)(ws + OFF_DENOM);
    float*  eArr   = (float*)(ws + OFF_EARR);
    float*  out    = (float*)d_out;

    hipLaunchKernelGGL(gat_init,       dim3(50000),  dim3(256), 0, stream, out, segmax, denom);
    hipLaunchKernelGGL(gat_pack_w,     dim3(128),    dim3(256), 0, stream, W, wpack);
    hipLaunchKernelGGL(gat_gemm,       dim3(M_TILES),dim3(256), 0, stream, x, wpack, a, h, ssrc, sdst);
    hipLaunchKernelGGL(gat_edge_score, dim3(6250),   dim3(256), 0, stream, src, dst, ssrc, sdst, eArr, segmax);
    hipLaunchKernelGGL(gat_aggregate,  dim3(200000), dim3(256), 0, stream, src, dst, eArr, segmax, h, denom, out);
    hipLaunchKernelGGL(gat_finalize,   dim3(50000),  dim3(256), 0, stream, out, denom);
}
